// SageEncoder_78735340470386
// MI455X (gfx1250) — matrix operand order repaired and checked by reading, not compiled, not run on hardware
//
#include <hip/hip_runtime.h>

#define BATCH 4096
#define F0 25
#define F1 10
#define FEAT 128
#define DIM 128

typedef __attribute__((ext_vector_type(16))) __bf16 v16bf;
typedef __attribute__((ext_vector_type(8)))  float  v8f;

union Frag { v16bf v; unsigned u[8]; };

__device__ __forceinline__ unsigned bf16pair(float a, float b) {
  unsigned ua = __float_as_uint(a);
  ua += 0x7FFFu + ((ua >> 16) & 1u);
  unsigned ub = __float_as_uint(b);
  ub += 0x7FFFu + ((ub >> 16) & 1u);
  return (ua >> 16) | (ub & 0xFFFF0000u);
}
__device__ __forceinline__ unsigned short bf16one(float a) {
  unsigned ua = __float_as_uint(a);
  ua += 0x7FFFu + ((ua >> 16) & 1u);
  return (unsigned short)(ua >> 16);
}
__device__ __forceinline__ float bflo(unsigned w) { return __uint_as_float(w << 16); }
__device__ __forceinline__ float bfhi(unsigned w) { return __uint_as_float(w & 0xFFFF0000u); }

// CDNA5 split-counter wait: intra-wave LDS RAW/WAR ordering + compiler barrier.
__device__ __forceinline__ void lds_fence() {
  asm volatile("s_wait_dscnt 0" ::: "memory");
}

// ---------------------------------------------------------------------------
// K1: repack [Ws;Wn] (256x128 fp32) per layer into bf16 WMMA B-fragments.
// Fragment order: wfrag[(((l*64 + kt*8 + nt)*32 + lane)*8 + r] (u32 = 2 bf16)
// B 32x16 bf16 layout: n = lane%16, k = kt*32 + (lane/16)*16 + 2r (+e)
// ---------------------------------------------------------------------------
__global__ __launch_bounds__(256) void sage_prep_weights(
    const float* __restrict__ Ws0, const float* __restrict__ Wn0,
    const float* __restrict__ Ws1, const float* __restrict__ Wn1,
    unsigned* __restrict__ wfrag) {
  const int t    = blockIdx.x * 256 + threadIdx.x;  // 0..32767
  const int r    = t & 7;
  const int lane = (t >> 3) & 31;
  const int nt   = (t >> 8) & 7;
  const int kt   = (t >> 11) & 7;
  const int l    = (t >> 14) & 1;
  const int n  = nt * 16 + (lane & 15);
  const int k0 = kt * 32 + (lane >> 4) * 16 + 2 * r;
  const int k1 = k0 + 1;
  const float* Wself  = l ? Ws1 : Ws0;
  const float* Wneigh = l ? Wn1 : Wn0;
  const float e0 = (k0 < 128) ? Wself[k0 * 128 + n] : Wneigh[(k0 - 128) * 128 + n];
  const float e1 = (k1 < 128) ? Wself[k1 * 128 + n] : Wneigh[(k1 - 128) * 128 + n];
  wfrag[t] = bf16pair(e0, e1);
}

// ---------------------------------------------------------------------------
// Wave-level GEMM: [16 x 256](LDS bf16) @ [256 x 128](pre-fragmented bf16)
// -> acc[8] (16x128 f32). A 16x32 bf16 layout: m = lane%16,
// k = kt*32 + (lane/16)*16 + 2i (+e), pairs contiguous.
// ---------------------------------------------------------------------------
__device__ __forceinline__ void wave_gemm(const unsigned* aw,
                                          const unsigned* __restrict__ wfrag_l,
                                          int lane, v8f acc[8]) {
  const int m    = lane & 15;
  const int half = lane >> 4;
#pragma unroll
  for (int kt = 0; kt < 8; ++kt) {
    Frag af;
#pragma unroll
    for (int i = 0; i < 8; ++i) {
      const int k = kt * 32 + half * 16 + i * 2;
      af.u[i] = aw[m * 128 + (k >> 1)];
    }
#pragma unroll
    for (int nt = 0; nt < 8; ++nt) {
      Frag bfr;
      const unsigned* bp = wfrag_l + (((kt * 8 + nt) * 32 + lane) * 8);
#pragma unroll
      for (int q = 0; q < 8; ++q) bfr.u[q] = bp[q];
      acc[nt] = __builtin_amdgcn_wmma_f32_16x16x32_bf16(
          false, af.v, false, bfr.v, (short)0, acc[nt], false, false);
    }
  }
}

// ---------------------------------------------------------------------------
// K2: hop-1 layer-0. 102400 rows; one 16-row tile per wave, 4 waves/WG.
// h1[r] = relu( [feat[ids1[r]] | mean_j feat[ids2[r*10+j]]] @ [Ws0;Wn0] )
// Output stored bf16 (halves layer-1 neighbor traffic).
// ---------------------------------------------------------------------------
__global__ __launch_bounds__(128) void sage_hop1(
    const int* __restrict__ ids1, const int* __restrict__ ids2,
    const float* __restrict__ feat, const unsigned* __restrict__ wfrag,
    unsigned short* __restrict__ h1) {
  __shared__ unsigned sA[4][2048];  // per-wave [16 rows][256 bf16] A-tile
  const int wave = threadIdx.x >> 5;
  const int lane = threadIdx.x & 31;
  const int r0 = (blockIdx.x * 4 + wave) * 16;
  unsigned* aw = sA[wave];

  for (int row = 0; row < 16; ++row) {
    const int r = r0 + row;
    const float4* sp = (const float4*)(feat + (size_t)ids1[r] * FEAT);
    const float4 v = sp[lane];                    // coalesced 512B row
    unsigned* dst = aw + row * 128;
    dst[lane * 2]     = bf16pair(v.x, v.y);
    dst[lane * 2 + 1] = bf16pair(v.z, v.w);
    float ax = 0.f, ay = 0.f, az = 0.f, aq = 0.f;
#pragma unroll
    for (int j = 0; j < F1; ++j) {
      const float4* np = (const float4*)(feat + (size_t)ids2[r * F1 + j] * FEAT);
      const float4 tv = np[lane];
      ax += tv.x; ay += tv.y; az += tv.z; aq += tv.w;
    }
    const float s = 1.0f / (float)F1;
    dst[64 + lane * 2]     = bf16pair(ax * s, ay * s);
    dst[64 + lane * 2 + 1] = bf16pair(az * s, aq * s);
  }
  lds_fence();

  v8f acc[8];
#pragma unroll
  for (int nt = 0; nt < 8; ++nt) acc[nt] = (v8f){0.f,0.f,0.f,0.f,0.f,0.f,0.f,0.f};
  wave_gemm(aw, wfrag, lane, acc);

#pragma unroll
  for (int nt = 0; nt < 8; ++nt) {
    const int n = nt * 16 + (lane & 15);
#pragma unroll
    for (int v2 = 0; v2 < 8; ++v2) {
      const int mm = (lane >> 4) * 8 + v2;
      float f = acc[nt][v2];
      f = f > 0.f ? f : 0.f;
      h1[(size_t)(r0 + mm) * DIM + n] = bf16one(f);
    }
  }
}

// ---------------------------------------------------------------------------
// K3: fused hop-0 layer-0 + layer-1. One 16-row output tile per wave.
//   h0  = relu([feat[ids0] | mean25 feat[ids1]] @ W0)      (WMMA GEMM #1)
//   out =      [h0          | mean25 h1        ] @ W1      (WMMA GEMM #2)
// ---------------------------------------------------------------------------
__global__ __launch_bounds__(128) void sage_out(
    const int* __restrict__ ids0, const int* __restrict__ ids1,
    const float* __restrict__ feat, const unsigned* __restrict__ wfrag,
    const unsigned* __restrict__ h1w, float* __restrict__ out) {
  __shared__ unsigned sA[4][2048];
  const int wave = threadIdx.x >> 5;
  const int lane = threadIdx.x & 31;
  const int b0 = (blockIdx.x * 4 + wave) * 16;
  unsigned* aw = sA[wave];

  // Stage layer-0 inputs for rows b0..b0+15.
  for (int row = 0; row < 16; ++row) {
    const int b = b0 + row;
    const float4* sp = (const float4*)(feat + (size_t)ids0[b] * FEAT);
    const float4 v = sp[lane];
    unsigned* dst = aw + row * 128;
    dst[lane * 2]     = bf16pair(v.x, v.y);
    dst[lane * 2 + 1] = bf16pair(v.z, v.w);
    float ax = 0.f, ay = 0.f, az = 0.f, aq = 0.f;
    for (int j = 0; j < F0; ++j) {
      const float4* np = (const float4*)(feat + (size_t)ids1[b * F0 + j] * FEAT);
      const float4 tv = np[lane];
      ax += tv.x; ay += tv.y; az += tv.z; aq += tv.w;
    }
    const float s = 1.0f / (float)F0;
    dst[64 + lane * 2]     = bf16pair(ax * s, ay * s);
    dst[64 + lane * 2 + 1] = bf16pair(az * s, aq * s);
  }
  lds_fence();

  v8f acc[8];
#pragma unroll
  for (int nt = 0; nt < 8; ++nt) acc[nt] = (v8f){0.f,0.f,0.f,0.f,0.f,0.f,0.f,0.f};
  wave_gemm(aw, wfrag, lane, acc);  // layer-0 fragments
  lds_fence();                      // GEMM#1 LDS reads done before overwrite

  // h0 (relu, bf16) -> self half of the A-tile.
  unsigned short* sh = (unsigned short*)aw;
#pragma unroll
  for (int nt = 0; nt < 8; ++nt) {
    const int n = nt * 16 + (lane & 15);
#pragma unroll
    for (int v2 = 0; v2 < 8; ++v2) {
      const int mm = (lane >> 4) * 8 + v2;
      float f = acc[nt][v2];
      f = f > 0.f ? f : 0.f;
      sh[mm * 256 + n] = bf16one(f);
    }
  }

  // mean of 25 bf16 h1 rows -> neighbor half of the A-tile.
  for (int row = 0; row < 16; ++row) {
    const int b = b0 + row;
    float a0 = 0.f, a1 = 0.f, a2 = 0.f, a3 = 0.f;
    for (int i = 0; i < F0; ++i) {
      const unsigned* hp = h1w + (size_t)(b * F0 + i) * (DIM / 2) + lane * 2;
      const unsigned w0 = hp[0], w1 = hp[1];
      a0 += bflo(w0); a1 += bfhi(w0); a2 += bflo(w1); a3 += bfhi(w1);
    }
    const float s = 1.0f / (float)F0;
    unsigned* dst = aw + row * 128;
    dst[64 + lane * 2]     = bf16pair(a0 * s, a1 * s);
    dst[64 + lane * 2 + 1] = bf16pair(a2 * s, a3 * s);
  }
  lds_fence();

  v8f acc2[8];
#pragma unroll
  for (int nt = 0; nt < 8; ++nt) acc2[nt] = (v8f){0.f,0.f,0.f,0.f,0.f,0.f,0.f,0.f};
  wave_gemm(aw, wfrag + 16384, lane, acc2);  // layer-1 fragments

#pragma unroll
  for (int nt = 0; nt < 8; ++nt) {
    const int n = nt * 16 + (lane & 15);
#pragma unroll
    for (int v2 = 0; v2 < 8; ++v2) {
      const int mm = (lane >> 4) * 8 + v2;
      out[(size_t)(b0 + mm) * DIM + n] = acc2[nt][v2];  // no activation
    }
  }
}

extern "C" void kernel_launch(void* const* d_in, const int* in_sizes, int n_in,
                              void* d_out, int out_size, void* d_ws, size_t ws_size,
                              hipStream_t stream) {
  const int*   ids0 = (const int*)d_in[0];
  const int*   ids1 = (const int*)d_in[1];
  const int*   ids2 = (const int*)d_in[2];
  const float* feat = (const float*)d_in[3];
  const float* Ws0  = (const float*)d_in[4];
  const float* Wn0  = (const float*)d_in[5];
  const float* Ws1  = (const float*)d_in[6];
  const float* Wn1  = (const float*)d_in[7];
  float* out = (float*)d_out;

  // Workspace layout: [0,128KB) bf16 weight fragments (both layers),
  // then 102400*128 bf16 h1 activations (~26.2 MB).
  unsigned*       wfrag = (unsigned*)d_ws;
  unsigned short* h1    = (unsigned short*)((char*)d_ws + 32768u * sizeof(unsigned));

  sage_prep_weights<<<128, 256, 0, stream>>>(Ws0, Wn0, Ws1, Wn1, wfrag);
  sage_hop1<<<(BATCH * F0) / 64, 128, 0, stream>>>(ids1, ids2, feat, wfrag, h1);
  sage_out<<<BATCH / 64, 128, 0, stream>>>(ids0, ids1, feat, wfrag,
                                           (const unsigned*)h1, out);
}
